// AttnBlock_9749575762835
// MI455X (gfx1250) — matrix operand order repaired and checked by reading, not compiled, not run on hardware
//
#include <hip/hip_runtime.h>
#include <math.h>

// ---------------------------------------------------------------------------
// Types for WMMA fragments (CDNA5 wave32: A/B = 16 halves/lane, C/D = 8 f32)
// ---------------------------------------------------------------------------
typedef __attribute__((ext_vector_type(16))) _Float16 v16h;
typedef __attribute__((ext_vector_type(8)))  _Float16 v8h;
typedef __attribute__((ext_vector_type(8)))  float    v8f;

union H16 { v16h v; v8h h[2]; };

__device__ __forceinline__ v8f wmma_f16(v16h a, v16h b, v8f c) {
  // D = A(16x32 f16) * B(32x16 f16) + C(16x16 f32)
  return __builtin_amdgcn_wmma_f32_16x16x32_f16(
      false, a, false, b, (short)0, c, false, false);
}

// Async global->LDS copy (CDNA5): per-lane 16B, tracked by ASYNCcnt.
// LDS flat addresses carry the workgroup-relative byte offset in addr[31:0].
__device__ __forceinline__ void async_ld_b128(void* lds_dst, const void* gsrc) {
  unsigned lofs = (unsigned)(uintptr_t)lds_dst;
  asm volatile("global_load_async_to_lds_b128 %0, %1, off"
               :: "v"(lofs), "v"(gsrc) : "memory");
}
__device__ __forceinline__ void wait_async() {
  asm volatile("s_wait_asynccnt 0" ::: "memory");
}

// Problem constants
#define BATCH   2
#define SEQ     4096          // 64*64
#define CH      512
#define HEADS   8
#define HD      64
#define GROUPS  32
#define GC      16            // channels per group
#define MROWS   (BATCH*SEQ)   // 8192

// ---------------------------------------------------------------------------
// 1) GroupNorm stats: one block per (b,g), reduce 4096*16 floats
// ---------------------------------------------------------------------------
__global__ __launch_bounds__(256)
void gn_stats_kernel(const float* __restrict__ x, float* __restrict__ stats) {
  int b = blockIdx.x >> 5;
  int g = blockIdx.x & 31;
  const float* base = x + (size_t)b * SEQ * CH + g * GC;
  float s = 0.f, ss = 0.f;
  for (int sp = threadIdx.x; sp < SEQ; sp += 256) {
    const float4* p = (const float4*)(base + (size_t)sp * CH);
#pragma unroll
    for (int j = 0; j < 4; ++j) {
      float4 v = p[j];
      s  += v.x + v.y + v.z + v.w;
      ss += v.x*v.x + v.y*v.y + v.z*v.z + v.w*v.w;
    }
  }
#pragma unroll
  for (int off = 16; off; off >>= 1) {
    s  += __shfl_down(s,  off, 32);
    ss += __shfl_down(ss, off, 32);
  }
  __shared__ float sh[16];
  int wave = threadIdx.x >> 5, lane = threadIdx.x & 31;
  if (lane == 0) { sh[wave] = s; sh[wave + 8] = ss; }
  __syncthreads();
  if (threadIdx.x == 0) {
    float ts = 0.f, tss = 0.f;
#pragma unroll
    for (int w = 0; w < 8; ++w) { ts += sh[w]; tss += sh[w + 8]; }
    float mean = ts * (1.f / 65536.f);
    float var  = tss * (1.f / 65536.f) - mean * mean;
    stats[blockIdx.x * 2 + 0] = mean;
    stats[blockIdx.x * 2 + 1] = rsqrtf(var + 1e-6f);
  }
}

// ---------------------------------------------------------------------------
// 2) Apply GroupNorm, cast to f16: h[M][512]
// ---------------------------------------------------------------------------
struct alignas(8) H4 { _Float16 a, b, c, d; };

__global__ __launch_bounds__(256)
void gn_apply_kernel(const float* __restrict__ x, const float* __restrict__ stats,
                     const float* __restrict__ scale, const float* __restrict__ bias,
                     _Float16* __restrict__ h) {
  size_t idx = ((size_t)blockIdx.x * 256 + threadIdx.x) * 4;
  int c = (int)(idx & (CH - 1));
  size_t row = idx >> 9;
  int b = (int)(row >> 12);
  int g = c >> 4;
  float mean = stats[(b * GROUPS + g) * 2 + 0];
  float rstd = stats[(b * GROUPS + g) * 2 + 1];
  float4 xv = *(const float4*)(x + idx);
  float4 sc = *(const float4*)(scale + c);
  float4 bi = *(const float4*)(bias + c);
  H4 o;
  o.a = (_Float16)((xv.x - mean) * rstd * sc.x + bi.x);
  o.b = (_Float16)((xv.y - mean) * rstd * sc.y + bi.y);
  o.c = (_Float16)((xv.z - mean) * rstd * sc.z + bi.z);
  o.d = (_Float16)((xv.w - mean) * rstd * sc.w + bi.w);
  *(H4*)(h + idx) = o;
}

// ---------------------------------------------------------------------------
// 3) Weight prep: wT[n][k] transposed f16, q-scale folded; bias; woT
// ---------------------------------------------------------------------------
__global__ __launch_bounds__(256)
void prep_wqkv_kernel(const float* __restrict__ wq, const float* __restrict__ wk,
                      const float* __restrict__ wv, _Float16* __restrict__ wT) {
  int idx = blockIdx.x * 256 + threadIdx.x;  // n*512 + k, n in [0,1536)
  int n = idx >> 9, k = idx & (CH - 1);
  int mat = n >> 9, nn = n & 511;
  const float* w = (mat == 0) ? wq : (mat == 1 ? wk : wv);
  float v = w[(size_t)k * 512 + nn];
  if (mat == 0) v *= 0.125f;  // hd^-0.5
  wT[idx] = (_Float16)v;
}

__global__ __launch_bounds__(256)
void prep_bias_kernel(const float* __restrict__ bq, const float* __restrict__ bk,
                      const float* __restrict__ bv, _Float16* __restrict__ bqkv) {
  int idx = blockIdx.x * 256 + threadIdx.x;
  if (idx < 1536) {
    int mat = idx >> 9, nn = idx & 511;
    float v = (mat == 0) ? bq[nn] * 0.125f : (mat == 1 ? bk[nn] : bv[nn]);
    bqkv[idx] = (_Float16)v;
  }
}

__global__ __launch_bounds__(256)
void prep_wo_kernel(const float* __restrict__ wo, _Float16* __restrict__ woT) {
  int idx = blockIdx.x * 256 + threadIdx.x;  // c*512 + hd
  int c = idx >> 9, hd = idx & 511;
  woT[idx] = (_Float16)wo[(size_t)hd * CH + c];
}

// ---------------------------------------------------------------------------
// 4) QKV GEMM: [8192 x 512] * [512 x 1536] -> q/k in [b*h][s][64],
//    V stored pre-TRANSPOSED as vT[b*h][d][s]. Block tile 128x64, 8 waves
//    each 32x32, K-step 32, double-buffered async global->LDS staging.
// ---------------------------------------------------------------------------
#define LDT 40   // padded LDS stride (halves)

__global__ __launch_bounds__(256)
void gemm_qkv_kernel(const _Float16* __restrict__ hmat, const _Float16* __restrict__ wT,
                     const _Float16* __restrict__ bqkv,
                     _Float16* __restrict__ q, _Float16* __restrict__ k,
                     _Float16* __restrict__ vT) {
  __shared__ _Float16 sA[2][128 * LDT];
  __shared__ _Float16 sB[2][64 * LDT];
  int m0 = blockIdx.x * 128;
  int n0 = blockIdx.y * 64;
  int tid = threadIdx.x, lane = tid & 31, wave = tid >> 5;
  int wm = wave & 3, wn = wave >> 2;
  int fm = lane & 15, g = lane >> 4;
  int arow = tid >> 1, ach = (tid & 1) * 16;   // 128 rows x 2 chunks of 16 halves
  v8f acc[2][2] = {};

  auto issue = [&](int kk, int buf) {
    const _Float16* asrc = hmat + (size_t)(m0 + arow) * CH + kk + ach;
    _Float16* adst = sA[buf] + arow * LDT + ach;
    async_ld_b128(adst,     asrc);
    async_ld_b128(adst + 8, asrc + 8);
    if (tid < 128) {
      const _Float16* bsrc = wT + (size_t)(n0 + arow) * CH + kk + ach;
      _Float16* bdst = sB[buf] + arow * LDT + ach;
      async_ld_b128(bdst,     bsrc);
      async_ld_b128(bdst + 8, bsrc + 8);
    }
  };

  issue(0, 0);
  for (int kk = 0; kk < CH; kk += 32) {
    int buf = (kk >> 5) & 1;
    wait_async();
    __syncthreads();
    if (kk + 32 < CH) issue(kk + 32, buf ^ 1);

    H16 af[2], bf[2];
#pragma unroll
    for (int i = 0; i < 2; ++i) {
      const _Float16* pa = sA[buf] + (wm * 32 + i * 16 + fm) * LDT;
      af[i].h[0] = *(const v8h*)(pa + g * 8);
      af[i].h[1] = *(const v8h*)(pa + 16 + g * 8);
      const _Float16* pb = sB[buf] + (wn * 32 + i * 16 + fm) * LDT;
      bf[i].h[0] = *(const v8h*)(pb + g * 8);
      bf[i].h[1] = *(const v8h*)(pb + 16 + g * 8);
    }
#pragma unroll
    for (int i = 0; i < 2; ++i)
#pragma unroll
      for (int j = 0; j < 2; ++j)
        acc[i][j] = wmma_f16(af[i].v, bf[j].v, acc[i][j]);
  }

  // Epilogue: +bias, scatter. q/k -> [bh][s][64]; v -> vT[bh][d][s]
#pragma unroll
  for (int i = 0; i < 2; ++i)
#pragma unroll
    for (int j = 0; j < 2; ++j) {
      int N = n0 + wn * 32 + j * 16 + fm;
      float bb = (float)bqkv[N];
      int mat = N >> 9, nn = N & 511;
      int head = nn >> 6, d = nn & 63;
#pragma unroll
      for (int r = 0; r < 8; ++r) {
        int M = m0 + wm * 32 + i * 16 + r + g * 8;
        int b_ = M >> 12, s = M & (SEQ - 1);
        float val = acc[i][j][r] + bb;
        _Float16 hv = (_Float16)val;
        int bh = b_ * HEADS + head;
        if (mat == 0)      q [(((size_t)bh * SEQ + s) << 6) + d] = hv;
        else if (mat == 1) k [(((size_t)bh * SEQ + s) << 6) + d] = hv;
        else               vT[(((size_t)bh * HD + d) << 12) + s] = hv;
      }
    }
}

// ---------------------------------------------------------------------------
// 5) Flash attention: block = (q-tile 128, bh), 8 waves x 16 q-rows, hd=64
//    K and Vt tiles double-buffered via async global->LDS (ASYNCcnt), so
//    tile t+1 DMA overlaps WMMA/softmax of tile t.
// ---------------------------------------------------------------------------
#define LDK 72   // padded LDS stride (halves)

__global__ __launch_bounds__(256)
void attn_kernel(const _Float16* __restrict__ q, const _Float16* __restrict__ kmat,
                 const _Float16* __restrict__ vT, _Float16* __restrict__ o) {
  __shared__ _Float16 sK[2][64 * LDK];     // [k][d]
  __shared__ _Float16 sVT[2][64 * LDK];    // [d][k]  (pre-transposed in HBM)
  __shared__ _Float16 sP[8][16 * LDK];     // per-wave P 16x64
  int qt = blockIdx.x, bh = blockIdx.y;
  int tid = threadIdx.x, lane = tid & 31, wave = tid >> 5;
  int fm = lane & 15, g = lane >> 4;
  size_t head_off = (size_t)bh * SEQ * HD;
  int qrow0 = qt * 128 + wave * 16;

  // Q A-fragments (held for whole kernel); rows are contiguous 64 halves
  H16 qa[2];
  {
    const _Float16* qp = q + head_off + (size_t)(qrow0 + fm) * HD;
#pragma unroll
    for (int ds = 0; ds < 2; ++ds) {
      qa[ds].h[0] = *(const v8h*)(qp + ds * 32 + g * 8);
      qa[ds].h[1] = *(const v8h*)(qp + ds * 32 + 16 + g * 8);
    }
  }

  float m_s[8], l_s[8];
#pragma unroll
  for (int r = 0; r < 8; ++r) { m_s[r] = -1e30f; l_s[r] = 0.f; }
  v8f oacc[4] = {};

  // coop-copy coordinates: 256 threads, 64 rows x 4 chunks of 16 halves (32B)
  int crow = tid >> 2, cch = (tid & 3) * 16;
  const _Float16* kbase = kmat + head_off;
  const _Float16* vbase = vT + head_off;

  auto issue_tile = [&](int kt, int buf) {
    const _Float16* ksrc = kbase + (size_t)(kt * 64 + crow) * HD + cch;
    _Float16* kdst = sK[buf] + crow * LDK + cch;
    async_ld_b128(kdst,     ksrc);
    async_ld_b128(kdst + 8, ksrc + 8);
    const _Float16* vsrc = vbase + ((size_t)crow << 12) + kt * 64 + cch;
    _Float16* vdst = sVT[buf] + crow * LDK + cch;
    async_ld_b128(vdst,     vsrc);
    async_ld_b128(vdst + 8, vsrc + 8);
  };

  issue_tile(0, 0);
  for (int kt = 0; kt < SEQ / 64; ++kt) {
    int buf = kt & 1;
    wait_async();
    __syncthreads();
    if (kt + 1 < SEQ / 64) issue_tile(kt + 1, buf ^ 1);

    // S = Q * K^T  (16 x 64 per wave)
    v8f sacc[4] = {};
#pragma unroll
    for (int ds = 0; ds < 2; ++ds)
#pragma unroll
      for (int nt = 0; nt < 4; ++nt) {
        H16 bf;
        const _Float16* pb = sK[buf] + (nt * 16 + fm) * LDK + ds * 32;
        bf.h[0] = *(const v8h*)(pb + g * 8); bf.h[1] = *(const v8h*)(pb + 16 + g * 8);
        sacc[nt] = wmma_f16(qa[ds].v, bf.v, sacc[nt]);
      }

    // Online softmax (rows at M = r + g*8; reduce across 16-lane halves)
#pragma unroll
    for (int r = 0; r < 8; ++r) {
      float mx = fmaxf(fmaxf(sacc[0][r], sacc[1][r]), fmaxf(sacc[2][r], sacc[3][r]));
#pragma unroll
      for (int off = 8; off; off >>= 1) mx = fmaxf(mx, __shfl_xor(mx, off, 32));
      float mn = fmaxf(m_s[r], mx);
      float alpha = __expf(m_s[r] - mn);
      float rs = 0.f;
#pragma unroll
      for (int nt = 0; nt < 4; ++nt) {
        float p = __expf(sacc[nt][r] - mn);
        sacc[nt][r] = p;
        rs += p;
      }
#pragma unroll
      for (int off = 8; off; off >>= 1) rs += __shfl_xor(rs, off, 32);
      l_s[r] = l_s[r] * alpha + rs;
      m_s[r] = mn;
#pragma unroll
      for (int nt = 0; nt < 4; ++nt) oacc[nt][r] *= alpha;
#pragma unroll
      for (int nt = 0; nt < 4; ++nt)
        sP[wave][(r + g * 8) * LDK + nt * 16 + fm] = (_Float16)sacc[nt][r];
    }

    // O += P * V  (sP is wave-private; DScnt keeps RAW order within the wave)
#pragma unroll
    for (int ks = 0; ks < 2; ++ks) {
      H16 pa;
      const _Float16* pp = &sP[wave][fm * LDK + ks * 32];
      pa.h[0] = *(const v8h*)(pp + g * 8);
      pa.h[1] = *(const v8h*)(pp + 16 + g * 8);
#pragma unroll
      for (int nt = 0; nt < 4; ++nt) {
        H16 bf;
        const _Float16* pb = sVT[buf] + (nt * 16 + fm) * LDK + ks * 32;
        bf.h[0] = *(const v8h*)(pb + g * 8); bf.h[1] = *(const v8h*)(pb + 16 + g * 8);
        oacc[nt] = wmma_f16(pa.v, bf.v, oacc[nt]);
      }
    }
  }

  // Epilogue: normalize, store o as [b][s][head*64+d] (GEMM-A friendly)
  int b_ = bh >> 3, head = bh & 7;
#pragma unroll
  for (int nt = 0; nt < 4; ++nt)
#pragma unroll
    for (int r = 0; r < 8; ++r) {
      int srow = qrow0 + r + g * 8;
      int d = nt * 16 + fm;
      float val = oacc[nt][r] / l_s[r];
      o[((size_t)(b_ * SEQ + srow) << 9) + head * 64 + d] = (_Float16)val;
    }
}

// ---------------------------------------------------------------------------
// 6) Output projection + residual: out = x + o[8192x512]*woT^T + bo  (f32)
// ---------------------------------------------------------------------------
__global__ __launch_bounds__(256)
void gemm_out_kernel(const _Float16* __restrict__ omat, const _Float16* __restrict__ woT,
                     const float* __restrict__ bo, const float* __restrict__ x,
                     float* __restrict__ out) {
  __shared__ _Float16 sA[2][128 * LDT];
  __shared__ _Float16 sB[2][64 * LDT];
  int m0 = blockIdx.x * 128;
  int n0 = blockIdx.y * 64;
  int tid = threadIdx.x, lane = tid & 31, wave = tid >> 5;
  int wm = wave & 3, wn = wave >> 2;
  int fm = lane & 15, g = lane >> 4;
  int arow = tid >> 1, ach = (tid & 1) * 16;
  v8f acc[2][2] = {};

  auto issue = [&](int kk, int buf) {
    const _Float16* asrc = omat + (size_t)(m0 + arow) * CH + kk + ach;
    _Float16* adst = sA[buf] + arow * LDT + ach;
    async_ld_b128(adst,     asrc);
    async_ld_b128(adst + 8, asrc + 8);
    if (tid < 128) {
      const _Float16* bsrc = woT + (size_t)(n0 + arow) * CH + kk + ach;
      _Float16* bdst = sB[buf] + arow * LDT + ach;
      async_ld_b128(bdst,     bsrc);
      async_ld_b128(bdst + 8, bsrc + 8);
    }
  };

  issue(0, 0);
  for (int kk = 0; kk < CH; kk += 32) {
    int buf = (kk >> 5) & 1;
    wait_async();
    __syncthreads();
    if (kk + 32 < CH) issue(kk + 32, buf ^ 1);

    H16 af[2], bf[2];
#pragma unroll
    for (int i = 0; i < 2; ++i) {
      const _Float16* pa = sA[buf] + (wm * 32 + i * 16 + fm) * LDT;
      af[i].h[0] = *(const v8h*)(pa + g * 8);
      af[i].h[1] = *(const v8h*)(pa + 16 + g * 8);
      const _Float16* pb = sB[buf] + (wn * 32 + i * 16 + fm) * LDT;
      bf[i].h[0] = *(const v8h*)(pb + g * 8);
      bf[i].h[1] = *(const v8h*)(pb + 16 + g * 8);
    }
#pragma unroll
    for (int i = 0; i < 2; ++i)
#pragma unroll
      for (int j = 0; j < 2; ++j)
        acc[i][j] = wmma_f16(af[i].v, bf[j].v, acc[i][j]);
  }

#pragma unroll
  for (int i = 0; i < 2; ++i)
#pragma unroll
    for (int j = 0; j < 2; ++j) {
      int N = n0 + wn * 32 + j * 16 + fm;
      float bb = bo[N];
#pragma unroll
      for (int r = 0; r < 8; ++r) {
        int M = m0 + wm * 32 + i * 16 + r + g * 8;
        size_t off = (size_t)M * CH + N;
        out[off] = x[off] + acc[i][j][r] + bb;
      }
    }
}

// ---------------------------------------------------------------------------
// Host launcher
// ---------------------------------------------------------------------------
extern "C" void kernel_launch(void* const* d_in, const int* in_sizes, int n_in,
                              void* d_out, int out_size, void* d_ws, size_t ws_size,
                              hipStream_t stream) {
  const float* x        = (const float*)d_in[0];
  const float* gn_scale = (const float*)d_in[1];
  const float* gn_bias  = (const float*)d_in[2];
  const float* wq       = (const float*)d_in[3];
  const float* bq       = (const float*)d_in[4];
  const float* wk       = (const float*)d_in[5];
  const float* bk       = (const float*)d_in[6];
  const float* wv       = (const float*)d_in[7];
  const float* bv       = (const float*)d_in[8];
  const float* wo       = (const float*)d_in[9];
  const float* bo       = (const float*)d_in[10];
  float* out = (float*)d_out;

  // Workspace layout (bytes), all 1KB aligned. Total ~34.0 MB.
  char* ws = (char*)d_ws;
  const size_t OFF_STATS = 0;                       //   512 B (64 * {mean,rstd})
  const size_t OFF_WT    = 1024;                    // 1536*512*2
  const size_t OFF_BQKV  = OFF_WT   + 1572864;      // 1536*2 (pad 4K)
  const size_t OFF_WOT   = OFF_BQKV + 4096;         // 512*512*2
  const size_t OFF_H     = OFF_WOT  + 524288;       // 8192*512*2 (reused as O)
  const size_t OFF_Q     = OFF_H    + 8388608;      // 16*4096*64*2
  const size_t OFF_K     = OFF_Q    + 8388608;
  const size_t OFF_V     = OFF_K    + 8388608;

  float*    stats = (float*)(ws + OFF_STATS);
  _Float16* wT    = (_Float16*)(ws + OFF_WT);
  _Float16* bqkv  = (_Float16*)(ws + OFF_BQKV);
  _Float16* woT   = (_Float16*)(ws + OFF_WOT);
  _Float16* hbuf  = (_Float16*)(ws + OFF_H);   // h, then reused for attention O
  _Float16* qbuf  = (_Float16*)(ws + OFF_Q);
  _Float16* kbuf  = (_Float16*)(ws + OFF_K);
  _Float16* vbuf  = (_Float16*)(ws + OFF_V);   // holds vT [bh][d][s]

  gn_stats_kernel<<<64, 256, 0, stream>>>(x, stats);
  gn_apply_kernel<<<4096, 256, 0, stream>>>(x, stats, gn_scale, gn_bias, hbuf);
  prep_wqkv_kernel<<<3072, 256, 0, stream>>>(wq, wk, wv, wT);
  prep_bias_kernel<<<6, 256, 0, stream>>>(bq, bk, bv, bqkv);
  prep_wo_kernel<<<1024, 256, 0, stream>>>(wo, woT);

  gemm_qkv_kernel<<<dim3(MROWS / 128, 1536 / 64), 256, 0, stream>>>(
      hbuf, wT, bqkv, qbuf, kbuf, vbuf);

  attn_kernel<<<dim3(SEQ / 128, BATCH * HEADS), 256, 0, stream>>>(
      qbuf, kbuf, vbuf, hbuf /* reuse as O buffer */);

  gemm_out_kernel<<<dim3(MROWS / 128, CH / 64), 256, 0, stream>>>(
      hbuf, woT, bo, x, out);
}
